// FocusMamba_27178553049270
// MI455X (gfx1250) — matrix operand order repaired and checked by reading, not compiled, not run on hardware
//
#include <hip/hip_runtime.h>
#include <hip/hip_bf16.h>

// FocusMamba forward for gfx1250 (MI455X, wave32, WMMA).
// All dense projections run through v_wmma_f32_16x16x32_f16 (f16 inputs,
// fp32 accumulate); scan/conv/LN are VALU kernels.
//
// GEMM fragment loads use index-clamping instead of predication: out-of-range
// A rows / B cols only affect C entries the epilogue never stores, so the
// K-loop is branch-free (uniform tail step handles K % 32 != 0).

typedef __attribute__((ext_vector_type(16))) _Float16 v16h;
typedef __attribute__((ext_vector_type(8)))  float    v8f;

#define NTOK   1444     // 4 frames * 19*19 tokens
#define EMB    384
#define DIN    768
#define XZW    1536     // 2*DIN
#define DBLW   56       // dt_rank + 2*d_state
#define DTRANK 24
#define NSTATE 16

// ---------------------------------------------------------------- utilities

__device__ __forceinline__ float silu_f(float x) { return x / (1.f + expf(-x)); }

__device__ __forceinline__ float apply_ep(float v, int mode) {
  if (mode == 1) {                       // softplus (jax.nn.softplus)
    return fmaxf(v, 0.f) + log1pf(expf(-fabsf(v)));
  }
  if (mode == 3) {                       // gelu (tanh approx, jax default)
    float c = v + 0.044715f * v * v * v;
    return 0.5f * v * (1.f + tanhf(0.7978845608028654f * c));
  }
  return v;                              // 0 = store, 2 = residual add
}

// ------------------------------------------------------- WMMA fragment I/O
// A fragment (16x32 f16): lane<16 -> row m, K chunks {0..7, 16..23};
//                         lane>=16 -> same rows, K chunks {8..15, 24..31}.
// p = row_base + k0 + hs*8 (32B aligned: all lda are multiples of 4 floats).
__device__ __forceinline__ v16h cvt_a(const float* __restrict__ p) {
  float4 u0 = *(const float4*)(p);
  float4 u1 = *(const float4*)(p + 4);
  float4 u2 = *(const float4*)(p + 16);
  float4 u3 = *(const float4*)(p + 20);
  v16h f;
  f[0]=(_Float16)u0.x;  f[1]=(_Float16)u0.y;  f[2]=(_Float16)u0.z;  f[3]=(_Float16)u0.w;
  f[4]=(_Float16)u1.x;  f[5]=(_Float16)u1.y;  f[6]=(_Float16)u1.z;  f[7]=(_Float16)u1.w;
  f[8]=(_Float16)u2.x;  f[9]=(_Float16)u2.y;  f[10]=(_Float16)u2.z; f[11]=(_Float16)u2.w;
  f[12]=(_Float16)u3.x; f[13]=(_Float16)u3.y; f[14]=(_Float16)u3.z; f[15]=(_Float16)u3.w;
  return f;
}

// B fragment (32x16 f16) from W[N,K] row-major (B = W^T):
// lane<16 -> col n, K chunks {0..7, 16..23}; lane>=16 -> K chunks {8..15, 24..31}.
// p = row_base + k0 + hs*8 (32B aligned).
__device__ __forceinline__ v16h cvt_b(const float* __restrict__ p) {
  float4 u0 = *(const float4*)(p);
  float4 u1 = *(const float4*)(p + 4);
  float4 u2 = *(const float4*)(p + 16);
  float4 u3 = *(const float4*)(p + 20);
  v16h f;
  f[0]=(_Float16)u0.x;  f[1]=(_Float16)u0.y;  f[2]=(_Float16)u0.z;  f[3]=(_Float16)u0.w;
  f[4]=(_Float16)u1.x;  f[5]=(_Float16)u1.y;  f[6]=(_Float16)u1.z;  f[7]=(_Float16)u1.w;
  f[8]=(_Float16)u2.x;  f[9]=(_Float16)u2.y;  f[10]=(_Float16)u2.z; f[11]=(_Float16)u2.w;
  f[12]=(_Float16)u3.x; f[13]=(_Float16)u3.y; f[14]=(_Float16)u3.z; f[15]=(_Float16)u3.w;
  return f;
}

// Checked K-tail loads (zero-fill k >= K); row = clamped row base pointer.
__device__ __forceinline__ v16h tail_a(const float* __restrict__ row, int hs,
                                       int k0, int K) {
  v16h f;
#pragma unroll
  for (int i = 0; i < 8; ++i) {
    int k1 = k0 + hs * 8 + i;
    int k2 = k0 + 16 + hs * 8 + i;
    f[i]     = (_Float16)(k1 < K ? row[k1] : 0.f);
    f[i + 8] = (_Float16)(k2 < K ? row[k2] : 0.f);
  }
  return f;
}

__device__ __forceinline__ v16h tail_b(const float* __restrict__ row, int hs,
                                       int k0, int K) {
  v16h f;
#pragma unroll
  for (int j = 0; j < 16; ++j) {
    int k = k0 + (j < 8 ? hs * 8 + j : 16 + hs * 8 + (j - 8));
    f[j] = (_Float16)(k < K ? row[k] : 0.f);
  }
  return f;
}

// --------------------------------------------------------------- GEMM core
// C[M,N](ldc) = epilogue(A[M,K](lda) @ W[N,K]^T(ldw) + bias); 8 waves/block
// in a 4x2 grid, each wave owns a 32x32 macro-tile (2x2 WMMA tiles).
__global__ __launch_bounds__(256) void wmma_gemm_k(
    const float* __restrict__ A, int lda,
    const float* __restrict__ W, int ldw,
    const float* __restrict__ bias,
    float* __restrict__ C, int ldc,
    int M, int N, int K, int mode) {
  int wave = threadIdx.x >> 5;
  int lane = threadIdx.x & 31;
  int wm = wave & 3, wn = wave >> 2;
  int m0 = blockIdx.x * 128 + wm * 32;
  int n0 = blockIdx.y * 64 + wn * 32;
  if (m0 >= M || n0 >= N) return;          // wave-uniform: EXEC stays full

  int ln16 = lane & 15, hs = lane >> 4;
  // Clamped per-lane row/col bases: OOB rows/cols produce garbage only in
  // C entries that the bounds-checked epilogue never writes.
  const float* ra0 = A + (long)min(m0 + ln16,      M - 1) * lda;
  const float* ra1 = A + (long)min(m0 + 16 + ln16, M - 1) * lda;
  const float* rb0 = W + (long)min(n0 + ln16,      N - 1) * ldw;
  const float* rb1 = W + (long)min(n0 + 16 + ln16, N - 1) * ldw;
  const float* pa0 = ra0 + hs * 8;
  const float* pa1 = ra1 + hs * 8;
  const float* pb0 = rb0 + hs * 8;
  const float* pb1 = rb1 + hs * 8;

  v8f acc[2][2] = {{{}, {}}, {{}, {}}};
  int Kfull = K & ~31;
  for (int k0 = 0; k0 < Kfull; k0 += 32) {
    v16h a0 = cvt_a(pa0 + k0);
    v16h a1 = cvt_a(pa1 + k0);
    v16h b0 = cvt_b(pb0 + k0);
    v16h b1 = cvt_b(pb1 + k0);
    acc[0][0] = __builtin_amdgcn_wmma_f32_16x16x32_f16(false, a0, false, b0, (short)0, acc[0][0], false, false);
    acc[0][1] = __builtin_amdgcn_wmma_f32_16x16x32_f16(false, a0, false, b1, (short)0, acc[0][1], false, false);
    acc[1][0] = __builtin_amdgcn_wmma_f32_16x16x32_f16(false, a1, false, b0, (short)0, acc[1][0], false, false);
    acc[1][1] = __builtin_amdgcn_wmma_f32_16x16x32_f16(false, a1, false, b1, (short)0, acc[1][1], false, false);
  }
  if (Kfull < K) {                         // uniform branch (K tail, e.g. K=24)
    v16h a0 = tail_a(ra0, hs, Kfull, K);
    v16h a1 = tail_a(ra1, hs, Kfull, K);
    v16h b0 = tail_b(rb0, hs, Kfull, K);
    v16h b1 = tail_b(rb1, hs, Kfull, K);
    acc[0][0] = __builtin_amdgcn_wmma_f32_16x16x32_f16(false, a0, false, b0, (short)0, acc[0][0], false, false);
    acc[0][1] = __builtin_amdgcn_wmma_f32_16x16x32_f16(false, a0, false, b1, (short)0, acc[0][1], false, false);
    acc[1][0] = __builtin_amdgcn_wmma_f32_16x16x32_f16(false, a1, false, b0, (short)0, acc[1][0], false, false);
    acc[1][1] = __builtin_amdgcn_wmma_f32_16x16x32_f16(false, a1, false, b1, (short)0, acc[1][1], false, false);
  }

#pragma unroll
  for (int tj = 0; tj < 2; ++tj) {
    int n = n0 + tj * 16 + ln16;
    if (n >= N) continue;
    float bv = bias ? bias[n] : 0.f;
#pragma unroll
    for (int ti = 0; ti < 2; ++ti) {
#pragma unroll
      for (int r = 0; r < 8; ++r) {
        int m = m0 + ti * 16 + r + 8 * hs;   // C/D layout: VGPR r, half hs
        if (m < M) {
          float v = apply_ep(acc[ti][tj][r] + bv, mode);
          float* cp = C + (long)m * ldc + n;
          if (mode == 2) *cp += v; else *cp = v;
        }
      }
    }
  }
}

// --------------------------------------------------------------- LayerNorm
__global__ __launch_bounds__(256) void ln_k(const float* __restrict__ X,
                                            float* __restrict__ Y,
                                            const float* __restrict__ g,
                                            const float* __restrict__ b, int ntok) {
  int wv = threadIdx.x >> 5, lane = threadIdx.x & 31;
  int tok = blockIdx.x * 8 + wv;
  if (tok >= ntok) return;
  const float* xr = X + (long)tok * EMB;
  float v[12];
  float s = 0.f;
#pragma unroll
  for (int i = 0; i < 12; ++i) { v[i] = xr[lane + 32 * i]; s += v[i]; }
#pragma unroll
  for (int o = 16; o >= 1; o >>= 1) s += __shfl_xor(s, o, 32);
  float mean = s * (1.f / 384.f);
  float q = 0.f;
#pragma unroll
  for (int i = 0; i < 12; ++i) { float d = v[i] - mean; q += d * d; }
#pragma unroll
  for (int o = 16; o >= 1; o >>= 1) q += __shfl_xor(q, o, 32);
  float inv = rsqrtf(q * (1.f / 384.f) + 1e-5f);
  float* yr = Y + (long)tok * EMB;
#pragma unroll
  for (int i = 0; i < 12; ++i) {
    int c = lane + 32 * i;
    yr[c] = (v[i] - mean) * inv * g[c] + b[c];
  }
}

// --------------------------------------------------- patch embed (14x14 s14)
__global__ __launch_bounds__(64) void patch_embed_k(
    const float* __restrict__ frames, const float* __restrict__ pw,
    const float* __restrict__ pb, float* __restrict__ x) {
  int tok = blockIdx.x;                         // 0..1443
  int co  = blockIdx.y * 64 + threadIdx.x;      // 0..383
  int t = tok / 361, hw = tok % 361;
  int gy = hw / 19, gx = hw % 19;
  const float mu[3]  = {0.485f, 0.456f, 0.406f};
  const float sd[3]  = {0.229f, 0.224f, 0.225f};
  const float* wr = pw + (long)co * 588;
  float acc = pb[co];
  int wi = 0;
  for (int ci = 0; ci < 3; ++ci) {
    float m = mu[ci], inv = 1.f / sd[ci];
    const float* img = frames + (((long)t * 3 + ci) * 266 + gy * 14) * 266 + gx * 14;
    for (int kh = 0; kh < 14; ++kh) {
      const float* row = img + kh * 266;
#pragma unroll
      for (int kw = 0; kw < 14; ++kw)
        acc += wr[wi++] * ((row[kw] - m) * inv);
    }
  }
  x[(long)tok * EMB + co] = acc;
}

// ------------------------- pos-embed bicubic 37x37 -> 19x19 (antialiased) + add
__device__ __forceinline__ float keys_cubic(float x) {
  x = fabsf(x);
  if (x < 1.f) return ((1.5f * x - 2.5f) * x) * x + 1.f;
  if (x < 2.f) return (((-0.5f * x + 2.5f) * x) - 4.f) * x + 2.f;
  return 0.f;
}

__global__ __launch_bounds__(256) void pos_add_k(const float* __restrict__ pos,
                                                 float* __restrict__ x) {
  int tid = blockIdx.x * 256 + threadIdx.x;
  if (tid >= 361 * EMB) return;
  int c = tid % EMB, hw = tid / EMB;
  int oy = hw / 19, ox = hw % 19;
  const float scale = 19.f / 37.f;              // downscale -> kernel widened
  float sy = (oy + 0.5f) / scale - 0.5f;
  float sx = (ox + 0.5f) / scale - 0.5f;
  float sumy = 0.f, sumx = 0.f;
  for (int i = 0; i < 37; ++i) {
    sumy += keys_cubic((sy - i) * scale);
    sumx += keys_cubic((sx - i) * scale);
  }
  float acc = 0.f;
  for (int iy = 0; iy < 37; ++iy) {
    float wy = keys_cubic((sy - iy) * scale);
    if (wy == 0.f) continue;
    const float* pr = pos + (long)(iy * 37) * EMB + c;
    float row = 0.f;
    for (int ix = 0; ix < 37; ++ix) {
      float wx = keys_cubic((sx - ix) * scale);
      row += wx * pr[(long)ix * EMB];
    }
    acc += wy * row;
  }
  acc /= (sumy * sumx);
  for (int t = 0; t < 4; ++t)
    x[(long)(t * 361 + hw) * EMB + c] += acc;
}

// ------------------------------------ causal depthwise conv (width 4) + SiLU
__global__ __launch_bounds__(256) void convsilu_k(
    const float* __restrict__ xz, const float* __restrict__ cw,
    const float* __restrict__ cb, float* __restrict__ xs,
    int nseq, int Lseq, int s_str, int l_str) {
  int tid = blockIdx.x * 256 + threadIdx.x;
  if (tid >= nseq * Lseq * DIN) return;
  int d = tid % DIN;
  int rest = tid / DIN;
  int l = rest % Lseq;
  int s = rest / Lseq;
  float acc = cb[d];
#pragma unroll
  for (int k = 0; k < 4; ++k) {
    int ll = l - 3 + k;
    if (ll >= 0) {
      long tok = s * s_str + ll * l_str;
      acc += cw[d * 4 + k] * xz[tok * XZW + d];
    }
  }
  long tok = s * s_str + l * l_str;
  xs[tok * DIN + d] = silu_f(acc);
}

// ------------------------------------------------ selective scan (S6) + gate
__global__ __launch_bounds__(256) void scan_k(
    const float* __restrict__ xz,   // gate z in cols [768,1536)
    const float* __restrict__ xs,   // conv+silu activations
    const float* __restrict__ dblp, // B at +24, C at +40 per token
    const float* __restrict__ dtp,  // softplus(dt), (ntok, 768)
    const float* __restrict__ A_log, const float* __restrict__ Dp,
    float* __restrict__ ys,
    int nseq, int Lseq, int s_str, int l_str) {
  int tid = blockIdx.x * 256 + threadIdx.x;
  if (tid >= nseq * DIN) return;
  int d = tid % DIN;
  int s = tid / DIN;
  float Ar[NSTATE], h[NSTATE];
#pragma unroll
  for (int j = 0; j < NSTATE; ++j) {
    Ar[j] = -expf(A_log[(long)d * NSTATE + j]);
    h[j] = 0.f;
  }
  float Dv = Dp[d];
  for (int l = 0; l < Lseq; ++l) {
    long tok = s * s_str + l * l_str;
    float dtv = dtp[tok * DIN + d];
    float xv  = xs[tok * DIN + d];
    const float* bc = dblp + tok * DBLW;
    float y = 0.f;
#pragma unroll
    for (int j = 0; j < NSTATE; ++j) {
      float dA = expf(dtv * Ar[j]);
      h[j] = dA * h[j] + dtv * bc[24 + j] * xv;
      y += h[j] * bc[40 + j];
    }
    y += Dv * xv;
    float zv = xz[tok * XZW + DIN + d];
    ys[tok * DIN + d] = y * silu_f(zv);
  }
}

// ------------------------------------------------------------------- driver
// jax pytree flattening (sorted dict keys):
//   0: frames
//   per block k (base = 1 + 32k):
//     +0..5  ffn:      b1, b2, norm_b, norm_g, w1, w2
//     +6..18 spatial:  A_log, D, conv_b, conv_w, dt_b, dt_w, in_b, in_w,
//                      norm_b, norm_g, out_b, out_w, xproj_w
//     +19..31 temporal: (same 13)
//   385: norm_b  386: norm_g  387: patch_b  388: patch_w  389: pos
extern "C" void kernel_launch(void* const* d_in, const int* in_sizes, int n_in,
                              void* d_out, int out_size, void* d_ws, size_t ws_size,
                              hipStream_t stream) {
  (void)in_sizes; (void)n_in; (void)out_size; (void)ws_size;
  auto in = [&](int i) { return (const float*)d_in[i]; };
  const float* frames  = in(0);
  const float* fin_nb  = in(385);
  const float* fin_ng  = in(386);
  const float* patch_b = in(387);
  const float* patch_w = in(388);
  const float* pos     = in(389);

  float* ws   = (float*)d_ws;
  float* x    = ws;                                 // NTOK*EMB
  float* xn   = x    + (size_t)NTOK * EMB;          // NTOK*EMB
  float* xz   = xn   + (size_t)NTOK * EMB;          // NTOK*1536
  float* xsil = xz   + (size_t)NTOK * XZW;          // NTOK*768
  float* dbl  = xsil + (size_t)NTOK * DIN;          // NTOK*56
  float* dtb  = dbl  + (size_t)NTOK * DBLW;         // NTOK*768
  float* ysb  = dtb  + (size_t)NTOK * DIN;          // NTOK*768

  auto gemm = [&](const float* A, int lda, const float* W, int ldw,
                  const float* bias, float* C, int ldc,
                  int M, int N, int K, int mode) {
    dim3 g((M + 127) / 128, (N + 63) / 64);
    wmma_gemm_k<<<g, 256, 0, stream>>>(A, lda, W, ldw, bias, C, ldc, M, N, K, mode);
  };

  patch_embed_k<<<dim3(NTOK, 6), 64, 0, stream>>>(frames, patch_w, patch_b, x);
  pos_add_k<<<(361 * EMB + 255) / 256, 256, 0, stream>>>(pos, x);

  int tap = 0;
  for (int blk = 0; blk < 12; ++blk) {
    int base = 1 + blk * 32;
    const float *f_b1 = in(base + 0), *f_b2 = in(base + 1),
                *f_nb = in(base + 2), *f_ng = in(base + 3),
                *f_w1 = in(base + 4), *f_w2 = in(base + 5);

    for (int m = 0; m < 2; ++m) {                   // 0: spatial, 1: temporal
      int mb = base + 6 + m * 13;
      const float *A_log = in(mb + 0),  *Dp     = in(mb + 1),
                  *conv_b = in(mb + 2), *conv_w = in(mb + 3),
                  *dt_b  = in(mb + 4),  *dt_w   = in(mb + 5),
                  *in_b  = in(mb + 6),  *in_w   = in(mb + 7),
                  *m_nb  = in(mb + 8),  *m_ng   = in(mb + 9),
                  *out_b = in(mb + 10), *out_w  = in(mb + 11),
                  *xproj_w = in(mb + 12);
      int nseq, Lseq, s_str, l_str;
      if (m == 0) { nseq = 4;   Lseq = 361; s_str = 361; l_str = 1;   }
      else        { nseq = 361; Lseq = 4;   s_str = 1;   l_str = 361; }

      ln_k<<<(NTOK + 7) / 8, 256, 0, stream>>>(x, xn, m_ng, m_nb, NTOK);
      gemm(xn, EMB, in_w, EMB, in_b, xz, XZW, NTOK, XZW, EMB, 0);       // in_proj
      int tot = nseq * Lseq * DIN;
      convsilu_k<<<(tot + 255) / 256, 256, 0, stream>>>(xz, conv_w, conv_b, xsil,
                                                        nseq, Lseq, s_str, l_str);
      gemm(xsil, DIN, xproj_w, DIN, nullptr, dbl, DBLW, NTOK, DBLW, DIN, 0); // x_proj
      gemm(dbl, DBLW, dt_w, DTRANK, dt_b, dtb, DIN, NTOK, DIN, DTRANK, 1);   // dt + softplus
      scan_k<<<(nseq * DIN + 255) / 256, 256, 0, stream>>>(xz, xsil, dbl, dtb,
                                                           A_log, Dp, ysb,
                                                           nseq, Lseq, s_str, l_str);
      gemm(ysb, DIN, out_w, DIN, out_b, x, EMB, NTOK, EMB, DIN, 2);     // out_proj + residual
    }

    // FFN
    ln_k<<<(NTOK + 7) / 8, 256, 0, stream>>>(x, xn, f_ng, f_nb, NTOK);
    gemm(xn, EMB, f_w1, EMB, f_b1, xz, XZW, NTOK, XZW, EMB, 3);         // gelu
    gemm(xz, XZW, f_w2, XZW, f_b2, x, EMB, NTOK, EMB, XZW, 2);          // residual

    if (blk == 2 || blk == 5 || blk == 8 || blk == 11) {
      ln_k<<<(NTOK + 7) / 8, 256, 0, stream>>>(
          x, (float*)d_out + (size_t)tap * NTOK * EMB, fin_ng, fin_nb, NTOK);
      ++tap;
    }
  }
}
